// PositionAttentionModule_5523327942726
// MI455X (gfx1250) — matrix operand order repaired and checked by reading, not compiled, not run on hardware
//
#include <hip/hip_runtime.h>

// Position Attention Module (DANet) for MI455X / gfx1250, wave32.
// Transposed flash attention:
//   S' = K Q^T   (M=j, N=i)   -> P' = exp2(S') in D-layout (Q pre-scaled
//                                by log2(e), so v_exp_f32 needs no mul)
//   O^T = V^T P' (M=c, N=i)   -> P' converts D-layout -> B-fragment with
//                                4 half-wave shuffles + v_cvt_pk_rtz_f16_f32
// Row sums via a 5th WMMA against an all-ones A matrix (no reductions).
// No running max: scores are O(1).
// K/V fragments are register double-buffered one 32-chunk ahead to hide
// L2 latency even at 1 wave/SIMD occupancy.

#define LQ 4096
#define CD 64
#define NB 4

typedef __fp16   fp16x2  __attribute__((ext_vector_type(2)));   // builtin's type
typedef _Float16 half8   __attribute__((ext_vector_type(8)));
typedef _Float16 half16  __attribute__((ext_vector_type(16)));
typedef float    float8  __attribute__((ext_vector_type(8)));

union H16 { half16 v; half8 h[2]; unsigned u[8]; };

#if __has_builtin(__builtin_amdgcn_exp2f)
#define EXP2F(x) __builtin_amdgcn_exp2f(x)
#else
extern "C" __device__ float __ocml_native_exp2_f32(float);
#define EXP2F(x) __ocml_native_exp2_f32(x)
#endif

// one v_cvt_pk_rtz_f16_f32: {lo: a, hi: b} as u32
static __device__ __forceinline__ unsigned pkrtz2(float a, float b) {
    union { fp16x2 h; unsigned u; } p;
    p.h = __builtin_amdgcn_cvt_pkrtz(a, b);
    return p.u;
}

// ---------------------------------------------------------------------------
// Kernel 1: Q = log2(e)*(wc@x+bc), K = wb@x+bb, V = wd@x+bd  (f16 outputs)
// Q,K stored (n, L, C) row-major; V stored (n, C, L) (natural layout).
// ---------------------------------------------------------------------------
__global__ __launch_bounds__(256) void qkv_kernel(
    const float* __restrict__ x,
    const float* __restrict__ wb, const float* __restrict__ bb,
    const float* __restrict__ wc, const float* __restrict__ bc,
    const float* __restrict__ wd, const float* __restrict__ bd,
    _Float16* __restrict__ Qh, _Float16* __restrict__ Kh,
    _Float16* __restrict__ Vh)
{
    int idx = blockIdx.x * blockDim.x + threadIdx.x;   // 0 .. NB*LQ-1
    int n = idx >> 12;          // LQ == 4096
    int l = idx & (LQ - 1);

    float xv[CD];
#pragma unroll
    for (int c = 0; c < CD; ++c)
        xv[c] = x[((size_t)n * CD + c) * LQ + l];

#pragma unroll 1
    for (int o = 0; o < CD; ++o) {
        float ak = bb[o], aq = bc[o], av = bd[o];
#pragma unroll
        for (int c = 0; c < CD; ++c) {
            float xc_ = xv[c];
            ak = fmaf(wb[o * CD + c], xc_, ak);   // keys
            aq = fmaf(wc[o * CD + c], xc_, aq);   // queries
            av = fmaf(wd[o * CD + c], xc_, av);   // values
        }
        Kh[((size_t)n * LQ + l) * CD + o] = (_Float16)ak;
        Qh[((size_t)n * LQ + l) * CD + o] = (_Float16)(aq * 1.44269504088896f);
        Vh[((size_t)n * CD + o) * LQ + l] = (_Float16)av;
    }
}

// ---------------------------------------------------------------------------
// Kernel 2: transposed flash attention.
// ---------------------------------------------------------------------------
struct Frags {          // K/V fragments for one 32-key chunk (64 VGPRs)
    H16 ka[2], kb[2];   // S' A-fragments: rows jb..jb+15 / jb+16..jb+31
    H16 vf[4];          // V^T A-fragments: 4 channel tiles
};

static __device__ __forceinline__ void load_frags(
    Frags& f, const _Float16* __restrict__ K, const _Float16* __restrict__ V,
    int jb, int ln, int g)
{
#pragma unroll
    for (int ks = 0; ks < 2; ++ks) {
        const _Float16* k0 = K + (size_t)(jb + ln) * CD + 16 * g + 32 * ks;
        const _Float16* k1 = K + (size_t)(jb + 16 + ln) * CD + 16 * g + 32 * ks;
        f.ka[ks].h[0] = *(const half8*)(k0);
        f.ka[ks].h[1] = *(const half8*)(k0 + 8);
        f.kb[ks].h[0] = *(const half8*)(k1);
        f.kb[ks].h[1] = *(const half8*)(k1 + 8);
    }
#pragma unroll
    for (int t = 0; t < 4; ++t) {
        const _Float16* vrow = V + (size_t)(16 * t + ln) * LQ + jb + 16 * g;
        f.vf[t].h[0] = *(const half8*)(vrow);
        f.vf[t].h[1] = *(const half8*)(vrow + 8);
    }
}

static __device__ __forceinline__ void attn_step(
    const Frags& f, const half16* qb, const H16& onesf,
    float8* oacc, float8& osum, int g)
{
    // ---- S' = K Q^T for two 16-j tiles
    float8 s0 = {}, s1 = {};
#pragma unroll
    for (int ks = 0; ks < 2; ++ks) {
        s0 = __builtin_amdgcn_wmma_f32_16x16x32_f16(false, f.ka[ks].v, false, qb[ks],
                                                    (short)0, s0, false, false);
        s1 = __builtin_amdgcn_wmma_f32_16x16x32_f16(false, f.kb[ks].v, false, qb[ks],
                                                    (short)0, s1, false, false);
    }

    // ---- P' = exp2(S')  (Q carries the log2(e) factor)
    float p0[8], p1[8];
#pragma unroll
    for (int r = 0; r < 8; ++r) {
        p0[r] = EXP2F(s0[r]);
        p1[r] = EXP2F(s1[r]);
    }

    // ---- D-layout -> B-fragment (K=j 32 x N=i 16).  Pack with
    //      v_cvt_pk_rtz_f16_f32, exchange the non-local half via lane^16.
    H16 pf;
#pragma unroll
    for (int q = 0; q < 4; ++q) {
        unsigned pk0  = pkrtz2(p0[2 * q], p0[2 * q + 1]);   // S0' rows 2q,2q+1 (+8g)
        unsigned pk1  = pkrtz2(p1[2 * q], p1[2 * q + 1]);   // S1' rows 2q,2q+1 (+8g)
        unsigned sent = g ? pk0 : pk1;                      // what the partner needs
        unsigned oth  = __shfl_xor(sent, 16, 32);
        pf.u[q]     = g ? oth : pk0;
        pf.u[4 + q] = g ? pk1 : oth;
    }

    // ---- O^T += V^T P' ; denominators via all-ones A
#pragma unroll
    for (int t = 0; t < 4; ++t)
        oacc[t] = __builtin_amdgcn_wmma_f32_16x16x32_f16(false, f.vf[t].v, false, pf.v,
                                                         (short)0, oacc[t], false, false);
    osum = __builtin_amdgcn_wmma_f32_16x16x32_f16(false, onesf.v, false, pf.v,
                                                  (short)0, osum, false, false);
}

__global__ __launch_bounds__(128) void attn_kernel(
    const _Float16* __restrict__ Qh, const _Float16* __restrict__ Kh,
    const _Float16* __restrict__ Vh, const float* __restrict__ x,
    const float* __restrict__ alphap, float* __restrict__ out)
{
    const int wave = threadIdx.x >> 5;
    const int lane = threadIdx.x & 31;
    const int g    = lane >> 4;     // half-wave group
    const int ln   = lane & 15;

    const int tile = blockIdx.x * 4 + wave;  // 0..1023
    const int n    = tile >> 8;              // 256 i-tiles per batch
    const int i0   = (tile & 255) << 4;      // first query of this wave

    const _Float16* Q = Qh + (size_t)n * LQ * CD;
    const _Float16* K = Kh + (size_t)n * LQ * CD;
    const _Float16* V = Vh + (size_t)n * CD * LQ;

    // Loop-invariant Q B-fragments (K=c x N=i; group 0 holds c 0..15 of the
    // 32-chunk, group 1 holds c 16..31, contiguous).
    half16 qb[2];
#pragma unroll
    for (int ks = 0; ks < 2; ++ks)
        qb[ks] = *(const half16*)(Q + (size_t)(i0 + ln) * CD + 32 * ks + 16 * g);

    H16 onesf;
#pragma unroll
    for (int q = 0; q < 8; ++q) onesf.u[q] = 0x3C003C00u;   // {1.0h, 1.0h}

    float8 oacc[4] = {};     // O^T tiles: c = 16t + r + 8g, i = i0 + ln
    float8 osum  = {};       // softmax denominators (all rows identical)

    // Ping-pong double buffer: fragments loaded one 32-chunk ahead.
    Frags fA, fB;
    load_frags(fA, K, V, 0, ln, g);
#pragma unroll 1
    for (int jb = 0; jb < LQ; jb += 64) {
        load_frags(fB, K, V, jb + 32, ln, g);
        attn_step(fA, qb, onesf, oacc, osum, g);
        load_frags(fA, K, V, (jb + 64) & (LQ - 1), ln, g);   // wraps on last iter
        attn_step(fB, qb, onesf, oacc, osum, g);
    }

    // ---- epilogue: out[n,c,i] = alpha * O^T[c,i]/l[i] + x[n,c,i]
    //      (lanes span i -> fully coalesced f32 stores)
    const float alpha = alphap[0];
    const float rinv  = 1.0f / osum[0];
#pragma unroll
    for (int t = 0; t < 4; ++t) {
#pragma unroll
        for (int r = 0; r < 8; ++r) {
            int c = 16 * t + r + 8 * g;
            size_t idx = ((size_t)n * CD + c) * LQ + i0 + ln;
            out[idx] = alpha * (oacc[t][r] * rinv) + x[idx];
        }
    }
}

// ---------------------------------------------------------------------------
extern "C" void kernel_launch(void* const* d_in, const int* in_sizes, int n_in,
                              void* d_out, int out_size, void* d_ws, size_t ws_size,
                              hipStream_t stream) {
    const float* x     = (const float*)d_in[0];
    const float* wb    = (const float*)d_in[1];
    const float* bb    = (const float*)d_in[2];
    const float* wc    = (const float*)d_in[3];
    const float* bc    = (const float*)d_in[4];
    const float* wd    = (const float*)d_in[5];
    const float* bd    = (const float*)d_in[6];
    const float* alpha = (const float*)d_in[7];

    _Float16* Qh = (_Float16*)d_ws;                    // 2 MiB
    _Float16* Kh = Qh + (size_t)NB * LQ * CD;          // 2 MiB
    _Float16* Vh = Kh + (size_t)NB * LQ * CD;          // 2 MiB

    qkv_kernel<<<dim3((NB * LQ) / 256), dim3(256), 0, stream>>>(
        x, wb, bb, wc, bc, wd, bd, Qh, Kh, Vh);

    attn_kernel<<<dim3((NB * LQ / 16) / 4), dim3(128), 0, stream>>>(
        Qh, Kh, Vh, x, alpha, (float*)d_out);
}
